// CausalSelfAttention_55336358643144
// MI455X (gfx1250) — compile-verified, matrix operand order checked by reading, not run on hardware
//
#include <hip/hip_runtime.h>
#include <hip/hip_bf16.h>

typedef __attribute__((ext_vector_type(16))) _Float16 v16h;
typedef __attribute__((ext_vector_type(8)))  float    v8f;

#define BDIM 4
#define SDIM 2048
#define EDIM 1024
#define HDIM 16
#define HD   64
#define JOINED 25

// ---------------------------------------------------------------------------
// WMMA 16-bit fragment gather, per CDNA5 ISA 7.12.2 layout:
//   lane L: row = L & 15 (M for A, N for B^T-stored tiles)
//   VGPR i holds K pair starting at: (i<4 ? 2i : 16+2(i-4)) + (L>=16 ? 8 : 0)
// Pairs are 16B-contiguous per lane -> compiler merges into ds_load_b128.
// ---------------------------------------------------------------------------
__device__ __forceinline__ int frag_klo(int i, int hi) {
    int base = (i < 4) ? (2 * i) : (16 + 2 * (i - 4));
    return base + (hi ? 8 : 0);
}

__device__ __forceinline__ v16h load_frag(const _Float16* __restrict__ p, int ld) {
    const int lane = threadIdx.x & 31;
    const int row  = lane & 15;
    const int hi   = lane >> 4;
    v16h r;
#pragma unroll
    for (int i = 0; i < 8; ++i) {
        int k = frag_klo(i, hi);
        r[2 * i]     = p[row * ld + k];
        r[2 * i + 1] = p[row * ld + k + 1];
    }
    return r;
}

// half-wave (16-lane) row reductions; wave32, masks < 16 stay within each half
__device__ __forceinline__ float hmax16(float v) {
#pragma unroll
    for (int m = 8; m; m >>= 1) v = fmaxf(v, __shfl_xor(v, m, 32));
    return v;
}
__device__ __forceinline__ float hsum16(float v) {
#pragma unroll
    for (int m = 8; m; m >>= 1) v += __shfl_xor(v, m, 32);
    return v;
}

// ---------------------------------------------------------------------------
// fp32 -> fp16 downconvert
// ---------------------------------------------------------------------------
__global__ void f32_to_f16_kernel(const float* __restrict__ src,
                                  _Float16* __restrict__ dst, int n) {
    int i = blockIdx.x * blockDim.x + threadIdx.x;
    int stride = gridDim.x * blockDim.x;
    for (; i < n; i += stride) dst[i] = (_Float16)src[i];
}

// ---------------------------------------------------------------------------
// Y = X @ W^T + b  (torch Linear). X:[M=8192][E] f16, W:[E][E] f16 row-major.
// Block tile 128x64, 8 waves in 4x2 layout; each wave owns a 32x32 register
// tile = 4 f32 accumulators -> 4 WMMAs per 4 fragment loads per K-step.
// Variadic macro so the epilogue block may contain commas.
// ---------------------------------------------------------------------------
#define GEMM_BODY(...)                                                         \
    __shared__ __align__(16) _Float16 As[128 * 32];                            \
    __shared__ __align__(16) _Float16 Bs[64 * 32];                             \
    const int m0   = blockIdx.x * 128;                                         \
    const int n0   = blockIdx.y * 64;                                          \
    const int tid  = threadIdx.x;                                              \
    const int wave = tid >> 5;                                                 \
    const int lane = tid & 31;                                                 \
    const int lo   = lane & 15;                                                \
    const int hi   = lane >> 4;                                                \
    const int wm   = wave >> 1; /* 0..3 : 32-row strip  */                     \
    const int wn   = wave & 1;  /* 0..1 : 32-col half   */                     \
    const int lrowA = tid >> 1;          /* 0..127 */                          \
    const int lcolA = (tid & 1) * 16;    /* 0,16 : 2x16B per thread */         \
    const int lrowB = tid >> 2;          /* 0..63  */                          \
    const int lcolB = (tid & 3) * 8;     /* 16B per thread */                  \
    v8f acc00 = {}, acc01 = {}, acc10 = {}, acc11 = {};                        \
    for (int k0 = 0; k0 < EDIM; k0 += 32) {                                    \
        __syncthreads();                                                       \
        *(uint4*)&As[lrowA * 32 + lcolA] =                                     \
            *(const uint4*)&X[(size_t)(m0 + lrowA) * EDIM + k0 + lcolA];       \
        *(uint4*)&As[lrowA * 32 + lcolA + 8] =                                 \
            *(const uint4*)&X[(size_t)(m0 + lrowA) * EDIM + k0 + lcolA + 8];   \
        *(uint4*)&Bs[lrowB * 32 + lcolB] =                                     \
            *(const uint4*)&W[(size_t)(n0 + lrowB) * EDIM + k0 + lcolB];       \
        if (k0 + 32 < EDIM)                                                    \
            __builtin_prefetch(&X[(size_t)(m0 + lrowA) * EDIM + k0 + 32 + lcolA], 0, 1); \
        __syncthreads();                                                       \
        v16h a0 = load_frag(&As[(wm * 32 + 0)  * 32], 32);                     \
        v16h a1 = load_frag(&As[(wm * 32 + 16) * 32], 32);                     \
        v16h b0 = load_frag(&Bs[(wn * 32 + 0)  * 32], 32);                     \
        v16h b1 = load_frag(&Bs[(wn * 32 + 16) * 32], 32);                     \
        acc00 = __builtin_amdgcn_wmma_f32_16x16x32_f16(false, a0, false, b0,   \
                                                       (short)0, acc00, false, false); \
        acc01 = __builtin_amdgcn_wmma_f32_16x16x32_f16(false, a0, false, b1,   \
                                                       (short)0, acc01, false, false); \
        acc10 = __builtin_amdgcn_wmma_f32_16x16x32_f16(false, a1, false, b0,   \
                                                       (short)0, acc10, false, false); \
        acc11 = __builtin_amdgcn_wmma_f32_16x16x32_f16(false, a1, false, b1,   \
                                                       (short)0, acc11, false, false); \
    }                                                                          \
    /* C/D layout: VGPR r -> row = r + 8*hi, col = lane&15 */                  \
    _Pragma("unroll")                                                          \
    for (int at = 0; at < 2; ++at) {                                           \
        _Pragma("unroll")                                                      \
        for (int bt = 0; bt < 2; ++bt) {                                       \
            v8f acc = (at == 0) ? ((bt == 0) ? acc00 : acc01)                  \
                                : ((bt == 0) ? acc10 : acc11);                 \
            const int f  = n0 + wn * 32 + bt * 16 + lo;                        \
            const float bv = bias[f];                                          \
            _Pragma("unroll")                                                  \
            for (int r = 0; r < 8; ++r) {                                      \
                const int gm = m0 + wm * 32 + at * 16 + r + 8 * hi;            \
                __VA_ARGS__                                                    \
            }                                                                  \
        }                                                                      \
    }

// QKV projection: f16 output, head-major [B][H][S][HD]
__global__ void __launch_bounds__(256)
gemm_qkv_kernel(const _Float16* __restrict__ X,
                const _Float16* __restrict__ W,
                const float*    __restrict__ bias,
                _Float16*       __restrict__ out) {
    GEMM_BODY(
        const int bb = gm / SDIM;
        const int ss = gm % SDIM;
        const int h = f >> 6;
        const int d = f & 63;
        out[(((size_t)bb * HDIM + h) * SDIM + ss) * HD + d] =
            (_Float16)(acc[r] + bv);
    )
}

// Output projection: fp32 row-major [B*S][E]
__global__ void __launch_bounds__(256)
gemm_proj_kernel(const _Float16* __restrict__ X,
                 const _Float16* __restrict__ W,
                 const float*    __restrict__ bias,
                 float*          __restrict__ out) {
    GEMM_BODY(
        out[(size_t)gm * EDIM + f] = acc[r] + bv;
    )
}

// ---------------------------------------------------------------------------
// Flash attention with periodic causal mask.
// Block = 4 waves; wave w owns 16 q-rows. Key blocks of 32 (WMMA K=32).
// 8 WMMAs per key block per wave (4 score + 4 PV); K/V tiles shared in LDS.
// ---------------------------------------------------------------------------
__global__ void __launch_bounds__(128)
flash_attn_kernel(const _Float16* __restrict__ Q,   // [B][H][S][HD]
                  const _Float16* __restrict__ K,
                  const _Float16* __restrict__ V,
                  _Float16*       __restrict__ O) { // [B][S][E]
    __shared__ __align__(16) _Float16 Ks[32 * 64];      // [key][d]
    __shared__ __align__(16) _Float16 Vs[64 * 32];      // [d][key] (transposed)
    __shared__ __align__(16) _Float16 Ps[4][16 * 32];   // per-wave P scratch

    const int qt = blockIdx.x, h = blockIdx.y, b = blockIdx.z;
    const int tid = threadIdx.x;
    const int wave = tid >> 5, lane = tid & 31;
    const int lo = lane & 15, hi = lane >> 4;

    const size_t head_off = ((size_t)b * HDIM + h) * SDIM * HD;
    const int qrow0 = qt * 64 + wave * 16;

    const _Float16* qp = Q + head_off + (size_t)qrow0 * HD;
    v16h qa0 = load_frag(qp + 0,  HD);   // hd 0..31
    v16h qa1 = load_frag(qp + 32, HD);   // hd 32..63

    v8f o0 = {}, o1 = {}, o2 = {}, o3 = {};
    float mrun[8], lrun[8];
#pragma unroll
    for (int r = 0; r < 8; ++r) { mrun[r] = -3.0e38f; lrun[r] = 0.0f; }

    const float scale = 0.125f;           // 1/sqrt(64)
    const int nkb = qt * 2 + 2;           // key blocks covering [0, qt*64+64)

    for (int kb = 0; kb < nkb; ++kb) {
        const int kbase = kb * 32;
        {   // stage K (row-major) and V (transposed) tiles
            const _Float16* kp = K + head_off + (size_t)kbase * HD;
            const _Float16* vp = V + head_off + (size_t)kbase * HD;
            int idx = tid * 16;
            *(uint4*)&Ks[idx]     = *(const uint4*)&kp[idx];
            *(uint4*)&Ks[idx + 8] = *(const uint4*)&kp[idx + 8];
#pragma unroll
            for (int j = 0; j < 16; ++j) {
                int e = idx + j;
                Vs[(e & 63) * 32 + (e >> 6)] = vp[e];
            }
        }
        __syncthreads();

        // scores: Q(16xHD) @ K^T(HDx32) -> two 16x16 tiles
        v8f s0 = {}, s1 = {};
        {
            v16h k0a = load_frag(&Ks[0  * 64 + 0],  64);
            v16h k0b = load_frag(&Ks[0  * 64 + 32], 64);
            v16h k1a = load_frag(&Ks[16 * 64 + 0],  64);
            v16h k1b = load_frag(&Ks[16 * 64 + 32], 64);
            s0 = __builtin_amdgcn_wmma_f32_16x16x32_f16(false, qa0, false, k0a,
                                                        (short)0, s0, false, false);
            s0 = __builtin_amdgcn_wmma_f32_16x16x32_f16(false, qa1, false, k0b,
                                                        (short)0, s0, false, false);
            s1 = __builtin_amdgcn_wmma_f32_16x16x32_f16(false, qa0, false, k1a,
                                                        (short)0, s1, false, false);
            s1 = __builtin_amdgcn_wmma_f32_16x16x32_f16(false, qa1, false, k1b,
                                                        (short)0, s1, false, false);
        }

        // online softmax (rows live in half-waves; hmax16/hsum16 reduce cols)
#pragma unroll
        for (int r = 0; r < 8; ++r) {
            const int qrow = qrow0 + r + 8 * hi;
            const int c0 = kbase + lo, c1 = kbase + 16 + lo;
            float v0 = s0[r] * scale;
            float v1 = s1[r] * scale;
            if (c0 > qrow || (c0 % JOINED) == (JOINED - 1)) v0 = -1.0e30f;
            if (c1 > qrow || (c1 % JOINED) == (JOINED - 1)) v1 = -1.0e30f;
            float mblk = hmax16(fmaxf(v0, v1));
            float mnew = fmaxf(mrun[r], mblk);
            float corr = __expf(mrun[r] - mnew);
            // guard: fully-masked entries contribute exactly zero
            float p0 = (v0 <= -1.0e29f) ? 0.0f : __expf(v0 - mnew);
            float p1 = (v1 <= -1.0e29f) ? 0.0f : __expf(v1 - mnew);
            lrun[r] = lrun[r] * corr + hsum16(p0 + p1);
            mrun[r] = mnew;
            o0[r] *= corr; o1[r] *= corr; o2[r] *= corr; o3[r] *= corr;
            int rl = r + 8 * hi;
            Ps[wave][rl * 32 + lo]      = (_Float16)p0;
            Ps[wave][rl * 32 + 16 + lo] = (_Float16)p1;
        }
        __syncthreads();

        // O += P(16x32) @ V(32x64): four 16x16 output tiles, K=32 keys
        {
            v16h pa  = load_frag(&Ps[wave][0], 32);
            v16h vb0 = load_frag(&Vs[0  * 32], 32);
            v16h vb1 = load_frag(&Vs[16 * 32], 32);
            v16h vb2 = load_frag(&Vs[32 * 32], 32);
            v16h vb3 = load_frag(&Vs[48 * 32], 32);
            o0 = __builtin_amdgcn_wmma_f32_16x16x32_f16(false, pa, false, vb0,
                                                        (short)0, o0, false, false);
            o1 = __builtin_amdgcn_wmma_f32_16x16x32_f16(false, pa, false, vb1,
                                                        (short)0, o1, false, false);
            o2 = __builtin_amdgcn_wmma_f32_16x16x32_f16(false, pa, false, vb2,
                                                        (short)0, o2, false, false);
            o3 = __builtin_amdgcn_wmma_f32_16x16x32_f16(false, pa, false, vb3,
                                                        (short)0, o3, false, false);
        }
        __syncthreads();
    }

    // epilogue: normalize and store row-major [B][S][E] f16
#pragma unroll
    for (int r = 0; r < 8; ++r) {
        const int qrow = qrow0 + r + 8 * hi;
        const float inv = 1.0f / lrun[r];
        const size_t rowoff = ((size_t)b * SDIM + qrow) * EDIM + h * HD;
        O[rowoff + 0  + lo] = (_Float16)(o0[r] * inv);
        O[rowoff + 16 + lo] = (_Float16)(o1[r] * inv);
        O[rowoff + 32 + lo] = (_Float16)(o2[r] * inv);
        O[rowoff + 48 + lo] = (_Float16)(o3[r] * inv);
    }
}

// ---------------------------------------------------------------------------
extern "C" void kernel_launch(void* const* d_in, const int* in_sizes, int n_in,
                              void* d_out, int out_size, void* d_ws, size_t ws_size,
                              hipStream_t stream) {
    const float* hs = (const float*)d_in[0];
    const float* Wq = (const float*)d_in[1];
    const float* bq = (const float*)d_in[2];
    const float* Wk = (const float*)d_in[3];
    const float* bk = (const float*)d_in[4];
    const float* Wv = (const float*)d_in[5];
    const float* bv = (const float*)d_in[6];
    const float* Wp = (const float*)d_in[7];
    const float* bp = (const float*)d_in[8];
    float* out = (float*)d_out;

    char* ws = (char*)d_ws;
    const size_t MB = 1u << 20;
    _Float16* Xh  = (_Float16*)(ws + 0 * MB);   // 16 MB : [B*S][E] f16
    _Float16* Wqh = (_Float16*)(ws + 16 * MB);  //  2 MB
    _Float16* Wkh = (_Float16*)(ws + 18 * MB);
    _Float16* Wvh = (_Float16*)(ws + 20 * MB);
    _Float16* Wph = (_Float16*)(ws + 22 * MB);
    _Float16* Qh  = (_Float16*)(ws + 24 * MB);  // 16 MB : [B][H][S][HD]
    _Float16* Kh  = (_Float16*)(ws + 40 * MB);
    _Float16* Vh  = (_Float16*)(ws + 56 * MB);
    _Float16* Oh  = (_Float16*)(ws + 72 * MB);  // 16 MB : [B][S][E]

    const int NX = BDIM * SDIM * EDIM;
    const int NW = EDIM * EDIM;
    f32_to_f16_kernel<<<4096, 256, 0, stream>>>(hs, Xh, NX);
    f32_to_f16_kernel<<<1024, 256, 0, stream>>>(Wq, Wqh, NW);
    f32_to_f16_kernel<<<1024, 256, 0, stream>>>(Wk, Wkh, NW);
    f32_to_f16_kernel<<<1024, 256, 0, stream>>>(Wv, Wvh, NW);
    f32_to_f16_kernel<<<1024, 256, 0, stream>>>(Wp, Wph, NW);

    dim3 gg((BDIM * SDIM) / 128, EDIM / 64);  // 64 x 16
    gemm_qkv_kernel<<<gg, 256, 0, stream>>>(Xh, Wqh, bq, Qh);
    gemm_qkv_kernel<<<gg, 256, 0, stream>>>(Xh, Wkh, bk, Kh);
    gemm_qkv_kernel<<<gg, 256, 0, stream>>>(Xh, Wvh, bv, Vh);

    dim3 ga(SDIM / 64, HDIM, BDIM);           // 32 x 16 x 4
    flash_attn_kernel<<<ga, 128, 0, stream>>>(Qh, Kh, Vh, Oh);

    gemm_proj_kernel<<<gg, 256, 0, stream>>>(Oh, Wph, bp, out);
}
